// GNN_ANF_14104672600360
// MI455X (gfx1250) — compile-verified, matrix operand order checked by reading, not run on hardware
//
#include <hip/hip_runtime.h>
#include <stdint.h>

// ---------------------------------------------------------------------------
// GNN forward (4 iterations) for MI455X / gfx1250.
//   - Dense GEMMs (MLPs, LSTM gates, readout) via v_wmma_f32_16x16x32_bf16
//     (bf16 inputs, f32 accumulate). Weights pre-packed into WMMA lane order
//     so each lane's B fragment is one contiguous 32-byte load.
//   - Edge-softmax attention, layernorm, LSTM state math in f32.
//   - ~96 MB of f32 scratch in d_ws (fits MI455X's 192 MB L2).
//   - All GEMM row counts are multiples of 16 (40000 / 10000), so tile loads
//     and stores are unguarded.
// ---------------------------------------------------------------------------

#define DEV __device__ __forceinline__

typedef __attribute__((ext_vector_type(16))) __bf16 v16bf;
typedef __attribute__((ext_vector_type(16))) unsigned short v16us;
typedef __attribute__((ext_vector_type(8))) unsigned short v8us;
typedef __attribute__((ext_vector_type(8))) float v8f;
static_assert(sizeof(v16bf) == 32 && sizeof(v16us) == 32, "vec size");

constexpr int D   = 64;      // feature dim
constexpr int LS  = 40000;   // l_size
constexpr int CH  = 10000;   // c_size/2
constexpr int HS  = 40000;   // h_size
constexpr int NB  = 64;      // batch
constexpr int NIT = 4;

DEV unsigned short f32_to_bf16(float f) {
  unsigned int u = __float_as_uint(f);
  unsigned int r = 0x7FFFu + ((u >> 16) & 1u);   // round-to-nearest-even
  return (unsigned short)((u + r) >> 16);
}

DEV float sigm(float x) { return 1.0f / (1.0f + expf(-x)); }

DEV v8f wmma_bf16(v16bf a, v16bf b, v8f c) {
  return __builtin_amdgcn_wmma_f32_16x16x32_bf16(false, a, false, b,
                                                 (short)0, c, false, false);
}

// --- WMMA tile loaders (ISA 7.12.2 layouts, wave32) ------------------------
// A (16x32 bf16): lanes 0-15 -> M=lane, K = k0+{0..7, 16..23};
//                 lanes 16-31 -> M=lane-16, K = k0+{8..15, 24..31}.
// Requires row0+16 <= n (callers guarantee n % 16 == 0).
DEV v16bf load_a_f32(const float* X, int row0, int k0) {
  int lane = threadIdx.x & 31;
  int m = row0 + (lane & 15);
  int kb = k0 + ((lane & 16) ? 8 : 0);
  const float4* p = (const float4*)(X + (size_t)m * D + kb);
  float v[16];
  *(float4*)(v + 0)  = p[0];
  *(float4*)(v + 4)  = p[1];
  *(float4*)(v + 8)  = p[4];   // +16 floats
  *(float4*)(v + 12) = p[5];
  v16us a;
#pragma unroll
  for (int i = 0; i < 16; ++i) a[i] = f32_to_bf16(v[i]);
  return __builtin_bit_cast(v16bf, a);
}

// A fragment from bf16(ushort) row-major LDS hidden [16 x ldk], 16B aligned.
DEV v16bf load_a_us(const unsigned short* X, int ldk, int k0) {
  int lane = threadIdx.x & 31;
  int m = lane & 15;
  int kb = k0 + ((lane & 16) ? 8 : 0);
  const unsigned short* p = X + m * ldk + kb;
  v8us lo = *(const v8us*)p;
  v8us hi = *(const v8us*)(p + 16);
  v16us a;
#pragma unroll
  for (int i = 0; i < 8; ++i) { a[i] = lo[i]; a[8 + i] = hi[i]; }
  return __builtin_bit_cast(v16bf, a);
}

// B fragment from pre-packed weights: [tile = nt*ktiles+kt][lane][16 halves].
DEV v16bf load_b_pk(const unsigned short* P, int ktiles, int kt, int nt) {
  int lane = threadIdx.x & 31;
  const v16us* p = (const v16us*)(P + (((size_t)(nt * ktiles + kt) * 32 + lane) << 4));
  return __builtin_bit_cast(v16bf, *p);
}

// --- small utility kernels -------------------------------------------------
__global__ void fill_kernel(float* p, float v, int n) {
  int i = blockIdx.x * 256 + threadIdx.x;
  if (i < n) p[i] = v;
}
__global__ void tile_row_kernel(const float* __restrict__ row, float* __restrict__ out, int total) {
  int i = blockIdx.x * 256 + threadIdx.x;
  if (i < total) out[i] = row[i & (D - 1)];
}
// Pack f32 row-major W[K][N] into bf16 WMMA-B lane order.
// Packed element tid: i = tid&15 (half idx), lane = (tid>>4)&31, tile = tid>>9,
// kt = tile % ktiles, nt = tile / ktiles.
__global__ void pack_b_kernel(const float* __restrict__ W, unsigned short* __restrict__ P,
                              int K, int N) {
  int tid = blockIdx.x * 256 + threadIdx.x;
  if (tid >= K * N) return;
  int i    = tid & 15;
  int lane = (tid >> 4) & 31;
  int tile = tid >> 9;
  int ktiles = K >> 5;
  int kt = tile % ktiles, nt = tile / ktiles;
  int n = nt * 16 + (lane & 15);
  int k = kt * 32 + ((lane & 16) ? 8 : 0) + (i & 7) + ((i & 8) ? 16 : 0);
  P[tid] = f32_to_bf16(W[(size_t)k * N + n]);
}

// --- segment-softmax attention (3 passes + init fills) ---------------------
DEV void atomicMaxF(float* addr, float v) {
  if (v >= 0.0f) atomicMax((int*)addr, __float_as_int(v));
  else           atomicMin((unsigned int*)addr, __float_as_uint(v));
}

// pass 1: per edge score s = dot(F[src],w)+b (masked -> -inf); segment max.
// 8 lanes per edge, 8 features per lane.
__global__ void att_score_kernel(const float* __restrict__ F, const int* __restrict__ src,
                                 const int* __restrict__ segr, int shift, int maskmode,
                                 const float* __restrict__ aw, const float* __restrict__ ab,
                                 int E, float* __restrict__ s_edge, float* __restrict__ seg_m) {
  int tid = blockIdx.x * 256 + threadIdx.x;
  int e = tid >> 3, sub = tid & 7;
  if (e >= E) return;
  int raw = segr[e];
  bool ok = (maskmode == 0) || ((maskmode == 1) ? ((raw & 1) == 0) : ((raw & 1) == 1));
  float part = 0.0f;
  if (ok) {
    const float4* f = (const float4*)(F + (size_t)src[e] * D + sub * 8);
    const float4* w = (const float4*)(aw + sub * 8);
    float4 f0 = f[0], f1 = f[1], w0 = w[0], w1 = w[1];
    part = f0.x * w0.x + f0.y * w0.y + f0.z * w0.z + f0.w * w0.w +
           f1.x * w1.x + f1.y * w1.y + f1.z * w1.z + f1.w * w1.w;
  }
  part += __shfl_xor(part, 1, 32);
  part += __shfl_xor(part, 2, 32);
  part += __shfl_xor(part, 4, 32);
  if (sub == 0) {
    float s = ok ? (part + ab[0]) : -__builtin_inff();
    s_edge[e] = s;
    if (ok) atomicMaxF(&seg_m[raw >> shift], s);
  }
}

// pass 2: e = exp(s - fix(m)); segment denominator.
__global__ void att_exp_kernel(const float* __restrict__ s_edge, const int* __restrict__ segr,
                               int shift, int E, const float* __restrict__ seg_m,
                               float* __restrict__ e_edge, float* __restrict__ seg_den) {
  int e = blockIdx.x * 256 + threadIdx.x;
  if (e >= E) return;
  float s = s_edge[e];
  float ev = 0.0f;
  if (s != -__builtin_inff()) {
    int seg = segr[e] >> shift;
    float m = seg_m[seg];
    if (!(m > -__builtin_inff())) m = 0.0f;   // reference: where(isfinite(m), m, 0)
    ev = expf(s - m);
    atomicAdd(&seg_den[seg], ev);
  }
  e_edge[e] = ev;
}

// pass 3: out[seg] += (e/den') * F[src]. 8 lanes per edge.
__global__ void att_scatter_kernel(const float* __restrict__ F, const int* __restrict__ src,
                                   const int* __restrict__ segr, int shift, int E,
                                   const float* __restrict__ e_edge, const float* __restrict__ seg_den,
                                   float* __restrict__ out) {
  int tid = blockIdx.x * 256 + threadIdx.x;
  int e = tid >> 3, sub = tid & 7;
  if (e >= E) return;
  float ev = e_edge[e];
  if (ev == 0.0f) return;
  int seg = segr[e] >> shift;
  float den = seg_den[seg];
  float w = ev / (den > 0.0f ? den : 1.0f);
  const float4* f = (const float4*)(F + (size_t)src[e] * D + sub * 8);
  float4 f0 = f[0], f1 = f[1];
  float* o = out + (size_t)seg * D + sub * 8;
  atomicAdd(&o[0], w * f0.x); atomicAdd(&o[1], w * f0.y);
  atomicAdd(&o[2], w * f0.z); atomicAdd(&o[3], w * f0.w);
  atomicAdd(&o[4], w * f1.x); atomicAdd(&o[5], w * f1.y);
  atomicAdd(&o[6], w * f1.z); atomicAdd(&o[7], w * f1.w);
}

// --- 2-layer MLP: Y[(m*ostride+ooff)] = relu(X@W1+b1)@W2+b2 ---------------
// W1 packed (K=64, N=H), W2 packed (K=H, N=64). n % 16 == 0.
template <int H>
__global__ void __launch_bounds__(128)
mlp2_kernel(const float* __restrict__ X, int n,
            const unsigned short* __restrict__ W1, const float* __restrict__ b1,
            const unsigned short* __restrict__ W2, const float* __restrict__ b2,
            float* __restrict__ Y, int ostride, int ooff) {
  __shared__ unsigned short Hid[16 * H];
  int row0 = blockIdx.x * 16;
  int wave = threadIdx.x >> 5;
  int lane = threadIdx.x & 31;
  for (int t = wave; t < H / 16; t += 4) {          // layer 1
    v8f c = {};
#pragma unroll
    for (int kt = 0; kt < 2; ++kt)
      c = wmma_bf16(load_a_f32(X, row0, kt * 32), load_b_pk(W1, 2, kt, t), c);
    int nn = lane & 15, mb = (lane & 16) ? 8 : 0;
    float bv = b1[t * 16 + nn];
#pragma unroll
    for (int r = 0; r < 8; ++r)
      Hid[(mb + r) * H + t * 16 + nn] = f32_to_bf16(fmaxf(c[r] + bv, 0.0f));
  }
  __syncthreads();
  {                                                 // layer 2 (64 cols)
    int t = wave;
    v8f c = {};
#pragma unroll
    for (int kt = 0; kt < H / 32; ++kt)
      c = wmma_bf16(load_a_us(Hid, H, kt * 32), load_b_pk(W2, H / 32, kt, t), c);
    int nn = lane & 15, mb = (lane & 16) ? 8 : 0;
    float bv = b2[t * 16 + nn];
#pragma unroll
    for (int r = 0; r < 8; ++r) {
      int m = row0 + mb + r;
      Y[((size_t)m * ostride + ooff) * D + t * 16 + nn] = c[r] + bv;
    }
  }
}

// --- layernorm-LSTM cell: gates = x@Wx + h@Wh, per-gate LN, state update ---
// Wx/Wh packed (K=64, N=256). n % 16 == 0.
__global__ void __launch_bounds__(256)
lstm_kernel(const float* __restrict__ Xin, float* __restrict__ Hst, float* __restrict__ Cst,
            int n, const unsigned short* __restrict__ Wx, const unsigned short* __restrict__ Wh,
            const float* __restrict__ lng, const float* __restrict__ lnb,
            const float* __restrict__ lncg, const float* __restrict__ lncb) {
  __shared__ float G[16 * 256];
  int row0 = blockIdx.x * 16;
  int wave = threadIdx.x >> 5;
  int lane = threadIdx.x & 31;
  for (int t = wave; t < 16; t += 8) {              // gate GEMM, N=256
    v8f c = {};
#pragma unroll
    for (int kt = 0; kt < 2; ++kt) {
      c = wmma_bf16(load_a_f32(Xin, row0, kt * 32), load_b_pk(Wx, 2, kt, t), c);
      c = wmma_bf16(load_a_f32(Hst, row0, kt * 32), load_b_pk(Wh, 2, kt, t), c);
    }
    int nn = lane & 15, mb = (lane & 16) ? 8 : 0;
#pragma unroll
    for (int r = 0; r < 8; ++r) G[(mb + r) * 256 + t * 16 + nn] = c[r];
  }
  __syncthreads();
  int f0 = lane * 2;                                // 2 features / lane
  for (int rr = wave; rr < 16; rr += 8) {
    int row = row0 + rr;
    float act[4][2];
#pragma unroll
    for (int g = 0; g < 4; ++g) {
      float x0 = G[rr * 256 + g * 64 + f0];
      float x1 = G[rr * 256 + g * 64 + f0 + 1];
      float s = x0 + x1, q = x0 * x0 + x1 * x1;
#pragma unroll
      for (int o = 16; o; o >>= 1) { s += __shfl_xor(s, o, 32); q += __shfl_xor(q, o, 32); }
      float mu = s * (1.0f / 64.0f);
      float rs = rsqrtf(fmaxf(q * (1.0f / 64.0f) - mu * mu, 0.0f) + 1e-5f);
      act[g][0] = (x0 - mu) * rs * lng[g * 64 + f0]     + lnb[g * 64 + f0];
      act[g][1] = (x1 - mu) * rs * lng[g * 64 + f0 + 1] + lnb[g * 64 + f0 + 1];
    }
    float i0 = sigm(act[0][0]), i1 = sigm(act[0][1]);
    float fg0 = sigm(act[1][0]), fg1 = sigm(act[1][1]);
    float g0 = tanhf(act[2][0]), g1 = tanhf(act[2][1]);
    float o0 = sigm(act[3][0]), o1 = sigm(act[3][1]);
    size_t base = (size_t)row * D + f0;
    float cn0 = fg0 * Cst[base]     + i0 * g0;
    float cn1 = fg1 * Cst[base + 1] + i1 * g1;
    float s = cn0 + cn1, q = cn0 * cn0 + cn1 * cn1;
#pragma unroll
    for (int o = 16; o; o >>= 1) { s += __shfl_xor(s, o, 32); q += __shfl_xor(q, o, 32); }
    float mu = s * (1.0f / 64.0f);
    float rs = rsqrtf(fmaxf(q * (1.0f / 64.0f) - mu * mu, 0.0f) + 1e-5f);
    Cst[base]     = cn0;
    Cst[base + 1] = cn1;
    Hst[base]     = o0 * tanhf((cn0 - mu) * rs * lncg[f0]     + lncb[f0]);
    Hst[base + 1] = o1 * tanhf((cn1 - mu) * rs * lncg[f0 + 1] + lncb[f0 + 1]);
  }
}

// --- readout MLP 64 -> 64(relu) -> 1. W1 packed (K=64, N=64). n%16==0. -----
__global__ void __launch_bounds__(128)
readout_kernel(const float* __restrict__ X, int n,
               const unsigned short* __restrict__ W1, const float* __restrict__ b1,
               const float* __restrict__ w2, const float* __restrict__ b2,
               float* __restrict__ logit) {
  __shared__ float Hd[16 * 64];
  int row0 = blockIdx.x * 16;
  int wave = threadIdx.x >> 5;
  int lane = threadIdx.x & 31;
  {
    int t = wave;
    v8f c = {};
#pragma unroll
    for (int kt = 0; kt < 2; ++kt)
      c = wmma_bf16(load_a_f32(X, row0, kt * 32), load_b_pk(W1, 2, kt, t), c);
    int nn = lane & 15, mb = (lane & 16) ? 8 : 0;
    float bv = b1[t * 16 + nn];
#pragma unroll
    for (int r = 0; r < 8; ++r) Hd[(mb + r) * 64 + t * 16 + nn] = fmaxf(c[r] + bv, 0.0f);
  }
  __syncthreads();
  int f0 = lane * 2;
  for (int rr = wave; rr < 16; rr += 4) {
    float p = Hd[rr * 64 + f0] * w2[f0] + Hd[rr * 64 + f0 + 1] * w2[f0 + 1];
#pragma unroll
    for (int o = 16; o; o >>= 1) p += __shfl_xor(p, o, 32);
    int row = row0 + rr;
    if (lane == 0) logit[row] = p + b2[0];
  }
}

// --- batch segment-mean + sigmoid ------------------------------------------
__global__ void batch_accum_kernel(const float* __restrict__ g, const float* __restrict__ ng,
                                   const int* __restrict__ c_batch, int ch, float* __restrict__ acc) {
  int i = blockIdx.x * 256 + threadIdx.x;
  if (i >= ch) return;
  int b = c_batch[2 * i];
  atomicAdd(&acc[b], g[i]);
  atomicAdd(&acc[NB + b], ng[i]);
  atomicAdd(&acc[2 * NB + b], 1.0f);
}
__global__ void batch_final_kernel(const float* __restrict__ acc, float* __restrict__ out) {
  int b = threadIdx.x;
  if (b >= NB) return;
  float cnt = acc[2 * NB + b];
  cnt = cnt > 0.0f ? cnt : 1.0f;
  float gm = acc[b] / cnt, ngm = acc[NB + b] / cnt;
  out[b] = 1.0f / (1.0f + expf(-(gm + ngm) * 0.5f));
}

// ===========================================================================
extern "C" void kernel_launch(void* const* d_in, const int* in_sizes, int n_in,
                              void* d_out, int out_size, void* d_ws, size_t ws_size,
                              hipStream_t stream) {
  (void)n_in; (void)out_size; (void)ws_size;
  // Params flattened as JAX pytree leaves (dict keys sorted) in d_in[0..54]:
  //  0/1 att_c2l.b/w  2/3 att_l2l  4/5 att_l2l_center  6/7 att_neg_l2c
  //  8/9 att_pos_l2c  10 c_neg_init 11 c_pos_init
  //  12-15 g_neg_readout{W1,b1,W2,b2}  16-19 g_readout  20 l_init
  //  21-26 l_update{Wh,Wx,ln_b,ln_cb,ln_cg,ln_g}
  //  27-30 neg_c2l{W1,b1,W2,b2}  31-36 neg_c_update  37-40 neg_l2c
  //  41-44 pos_c2l  45-50 pos_c_update  51-54 pos_l2c
  //  55 l_edge 56 c_edge 57 l1_edge 58 l2_edge 59 c_batch 60-63 scalars
  auto Fp = [&](int i) { return (const float*)d_in[i]; };
  auto Ip = [&](int i) { return (const int*)d_in[i]; };
  const int E = in_sizes[55];

  const int* l_edge  = Ip(55);
  const int* c_edge  = Ip(56);
  const int* l1_edge = Ip(57);
  const int* l2_edge = Ip(58);
  const int* c_batch = Ip(59);

  // --- workspace carve-out (256B aligned) ---
  uintptr_t cur = (uintptr_t)d_ws;
  auto allocF = [&](size_t nel) {
    uintptr_t p = cur; cur += ((nel * 4 + 255) & ~(size_t)255); return (float*)p;
  };
  auto allocU = [&](size_t nel) {
    uintptr_t p = cur; cur += ((nel * 2 + 255) & ~(size_t)255); return (unsigned short*)p;
  };
  float* l_emb      = allocF((size_t)LS * D);
  float* l_state    = allocF((size_t)LS * D);
  float* pos_c_emb  = allocF((size_t)CH * D);
  float* neg_c_emb  = allocF((size_t)CH * D);
  float* pos_c_st   = allocF((size_t)CH * D);
  float* neg_c_st   = allocF((size_t)CH * D);
  float* l_center   = allocF((size_t)HS * D);
  float* c_msg      = allocF((size_t)2 * CH * D);
  float* c2l_aggr   = allocF((size_t)HS * D);
  float* l2l_aggr   = allocF((size_t)LS * D);
  float* pos_l_feat = allocF((size_t)HS * D);
  float* neg_l_feat = allocF((size_t)HS * D);
  float* pos_aggr   = allocF((size_t)CH * D);
  float* neg_aggr   = allocF((size_t)CH * D);
  float* s_edge     = allocF(E);
  float* e_edge     = allocF(E);
  float* seg_m      = allocF(HS);
  float* seg_den    = allocF(HS);
  float* g_logit    = allocF(CH);
  float* ng_logit   = allocF(CH);
  float* acc        = allocF(3 * NB);

  // --- pack GEMM weights to bf16 WMMA-B order once per call ---
  auto pack = [&](int idx, int K, int N) {
    unsigned short* p = allocU((size_t)K * N);
    pack_b_kernel<<<(K * N + 255) / 256, 256, 0, stream>>>(Fp(idx), p, K, N);
    return p;
  };
  unsigned short* Wc2lP1 = pack(41, 64, 64);   unsigned short* Wc2lP2 = pack(43, 64, 64);
  unsigned short* Wc2lN1 = pack(27, 64, 64);   unsigned short* Wc2lN2 = pack(29, 64, 64);
  unsigned short* Wl2cP1 = pack(51, 64, 128);  unsigned short* Wl2cP2 = pack(53, 128, 64);
  unsigned short* Wl2cN1 = pack(37, 64, 128);  unsigned short* Wl2cN2 = pack(39, 128, 64);
  unsigned short* WhL = pack(21, 64, 256);     unsigned short* WxL = pack(22, 64, 256);
  unsigned short* WhN = pack(31, 64, 256);     unsigned short* WxN = pack(32, 64, 256);
  unsigned short* WhP = pack(45, 64, 256);     unsigned short* WxP = pack(46, 64, 256);
  unsigned short* WrP = pack(16, 64, 64);      unsigned short* WrN = pack(12, 64, 64);

  auto fill = [&](float* p, float v, int nel) {
    fill_kernel<<<(nel + 255) / 256, 256, 0, stream>>>(p, v, nel);
  };
  auto attention = [&](const float* Feat, const int* src, const int* segr, int shift,
                       int maskmode, const float* aw, const float* ab, int nseg, float* out) {
    fill(out, 0.0f, nseg * D);
    fill(seg_m, -__builtin_inff(), nseg);
    fill(seg_den, 0.0f, nseg);
    int nt = E * 8;
    att_score_kernel<<<(nt + 255) / 256, 256, 0, stream>>>(Feat, src, segr, shift, maskmode,
                                                           aw, ab, E, s_edge, seg_m);
    att_exp_kernel<<<(E + 255) / 256, 256, 0, stream>>>(s_edge, segr, shift, E, seg_m,
                                                        e_edge, seg_den);
    att_scatter_kernel<<<(nt + 255) / 256, 256, 0, stream>>>(Feat, src, segr, shift, E,
                                                             e_edge, seg_den, out);
  };

  // --- init embeddings / states ---
  tile_row_kernel<<<(LS * D + 255) / 256, 256, 0, stream>>>(Fp(20), l_emb, LS * D);
  tile_row_kernel<<<(CH * D + 255) / 256, 256, 0, stream>>>(Fp(11), pos_c_emb, CH * D);
  tile_row_kernel<<<(CH * D + 255) / 256, 256, 0, stream>>>(Fp(10), neg_c_emb, CH * D);
  fill(l_state, 0.0f, LS * D);
  fill(pos_c_st, 0.0f, CH * D);
  fill(neg_c_st, 0.0f, CH * D);

  // --- message-passing iterations ---
  for (int it = 0; it < NIT; ++it) {
    // l_emb_center = att(l_emb[l1], seg=l2) over h_size
    attention(l_emb, l1_edge, l2_edge, 0, 0, Fp(5), Fp(4), HS, l_center);
    // clause features (interleaved pos/neg rows of c_msg)
    mlp2_kernel<64><<<CH / 16, 128, 0, stream>>>(pos_c_emb, CH, Wc2lP1, Fp(42), Wc2lP2, Fp(44), c_msg, 2, 0);
    mlp2_kernel<64><<<CH / 16, 128, 0, stream>>>(neg_c_emb, CH, Wc2lN1, Fp(28), Wc2lN2, Fp(30), c_msg, 2, 1);
    // c2l_aggr = att(c_msg[c_edge], seg=l_edge) over h_size
    attention(c_msg, c_edge, l_edge, 0, 0, Fp(1), Fp(0), HS, c2l_aggr);
    // l2l_aggr = att(c2l_aggr[l2_edge], seg=l1_edge) over l_size
    attention(c2l_aggr, l2_edge, l1_edge, 0, 0, Fp(3), Fp(2), LS, l2l_aggr);
    // literal LSTM update
    lstm_kernel<<<LS / 16, 256, 0, stream>>>(l2l_aggr, l_emb, l_state, LS, WxL, WhL,
                                             Fp(26), Fp(23), Fp(25), Fp(24));
    // literal->clause features from (pre-update) l_emb_center
    mlp2_kernel<128><<<HS / 16, 128, 0, stream>>>(l_center, HS, Wl2cP1, Fp(52), Wl2cP2, Fp(54), pos_l_feat, 1, 0);
    mlp2_kernel<128><<<HS / 16, 128, 0, stream>>>(l_center, HS, Wl2cN1, Fp(38), Wl2cN2, Fp(40), neg_l_feat, 1, 0);
    // masked attention into half-clauses (seg = c_edge>>1, parity mask)
    attention(pos_l_feat, l_edge, c_edge, 1, 1, Fp(9), Fp(8), CH, pos_aggr);
    attention(neg_l_feat, l_edge, c_edge, 1, 2, Fp(7), Fp(6), CH, neg_aggr);
    // clause LSTM updates
    lstm_kernel<<<CH / 16, 256, 0, stream>>>(pos_aggr, pos_c_emb, pos_c_st, CH, WxP, WhP,
                                             Fp(50), Fp(47), Fp(49), Fp(48));
    lstm_kernel<<<CH / 16, 256, 0, stream>>>(neg_aggr, neg_c_emb, neg_c_st, CH, WxN, WhN,
                                             Fp(36), Fp(33), Fp(35), Fp(34));
  }

  // --- readout + batch mean + sigmoid ---
  readout_kernel<<<CH / 16, 128, 0, stream>>>(pos_c_emb, CH, WrP, Fp(17), Fp(18), Fp(19), g_logit);
  readout_kernel<<<CH / 16, 128, 0, stream>>>(neg_c_emb, CH, WrN, Fp(13), Fp(14), Fp(15), ng_logit);
  fill(acc, 0.0f, 3 * NB);
  batch_accum_kernel<<<(CH + 255) / 256, 256, 0, stream>>>(g_logit, ng_logit, c_batch, CH, acc);
  batch_final_kernel<<<1, NB, 0, stream>>>(acc, (float*)d_out);
}
